// Channel_Attention_83502754169335
// MI455X (gfx1250) — matrix operand order repaired and checked by reading, not compiled, not run on hardware
//
#include <hip/hip_runtime.h>
#include <hip/hip_bf16.h>
#include <math.h>

typedef __attribute__((ext_vector_type(16))) __bf16 v16bf;
typedef __attribute__((ext_vector_type(8)))  __bf16 v8bf;
typedef __attribute__((ext_vector_type(4)))  __bf16 v4bf;
typedef __attribute__((ext_vector_type(8)))  float  v8f;

#define HEADS 8
#define PS    7
#define CDIM  384
#define HDIM  224
#define WDIM  224
#define HW    (HDIM*WDIM)   // 50176
#define DHEAD 48
#define NW    32
#define WIN   49

// low 32 bits of a flat LDS pointer == LDS byte offset (aperture in [63:32])
__device__ __forceinline__ unsigned lds_off(const void* p) {
    return (unsigned)(unsigned long long)p;
}

// ============================================================
// f32 -> bf16 bulk convert (vectorized 4-wide)
// ============================================================
__global__ __launch_bounds__(256)
void f32_to_bf16(const float* __restrict__ in, __bf16* __restrict__ out, int n4)
{
    int i = blockIdx.x * 256 + threadIdx.x;
    if (i >= n4) return;
    float4 v = ((const float4*)in)[i];
    v4bf o;
    o[0] = (__bf16)v.x; o[1] = (__bf16)v.y; o[2] = (__bf16)v.z; o[3] = (__bf16)v.w;
    ((v4bf*)out)[i] = o;
}

// ============================================================
// GEMM:  Y[b][o][p] = sum_c Wm[o][c] * X[b][c][p]   (bf16 x bf16 -> OutT)
// Block tile 128x64, K steps of 32, 8 waves (4x2), 2x2 wmma frags/wave.
// A tile: async DMA global->LDS (no conversion needed, bf16 weights).
// B tile: b128 global loads + transpose scatter into LDS.
// ============================================================
#define GT_M 128
#define GT_N 64
#define GT_K 32
#define LDA  40   // bf16 elems per row: 80 B stride, 16B aligned

template <typename OutT>
__global__ __launch_bounds__(256)
void gemm_wmma_bf16(const __bf16* __restrict__ Wm,  // [O x K] bf16
                    const __bf16* __restrict__ X,   // [batch][K x P] bf16
                    OutT* __restrict__ Y,           // [batch][O x P]
                    int O, int K, int P)
{
    __shared__ __bf16 sA[GT_M * LDA];   // A row-major [m][k]
    __shared__ __bf16 sB[GT_N * LDA];   // B transposed [n][k]

    const int tid   = threadIdx.x;
    const int mBase = blockIdx.y * GT_M;
    const int nBase = blockIdx.x * GT_N;
    const __bf16* Xb = X + (size_t)blockIdx.z * (size_t)K * (size_t)P;
    OutT*         Yb = Y + (size_t)blockIdx.z * (size_t)O * (size_t)P;

    const int wid = tid >> 5;
    const int ln  = tid & 31;
    const int lr  = ln & 15;
    const int lhi = ln >> 4;
    const int wm0 = (wid >> 1) * 32;
    const int wn0 = (wid & 1)  * 32;

    v8f acc[2][2];
#pragma unroll
    for (int mi = 0; mi < 2; mi++)
#pragma unroll
        for (int nj = 0; nj < 2; nj++)
#pragma unroll
            for (int r = 0; r < 8; r++) acc[mi][nj][r] = 0.0f;

    for (int k0 = 0; k0 < K; k0 += GT_K) {
        __syncthreads();
        // ---- A tile 128x32 bf16 (8 KB): async DMA to LDS, 2 x b128 per lane
#pragma unroll
        for (int i = 0; i < 2; i++) {
            int c  = i * 256 + tid;          // 512 chunks of 16 B
            int m  = c >> 2;                 // 4 chunks per 64 B row
            int kc = (c & 3) * 8;            // bf16 element offset in row
            unsigned lao = lds_off(&sA[m * LDA + kc]);
            unsigned gvo = (unsigned)(((mBase + m) * K + k0 + kc) * 2);  // bytes
            asm volatile("global_load_async_to_lds_b128 %0, %1, %2"
                         :: "v"(lao), "v"(gvo), "s"(Wm) : "memory");
        }
        // ---- B tile 32x64 bf16: b128 load + transpose scatter
        {
            int k  = tid >> 3;               // 0..31
            int n0 = (tid & 7) * 8;          // 0..56
            v8bf bv = *(const v8bf*)(Xb + (size_t)(k0 + k) * P + (nBase + n0));
#pragma unroll
            for (int j = 0; j < 8; j++)
                sB[(n0 + j) * LDA + k] = bv[j];
        }
        asm volatile("s_wait_asynccnt 0x0" ::: "memory");
        __syncthreads();

        // ---- fragments (CDNA5 ISA 7.12.2 layouts) + 4 WMMA
        v16bf a[2], bf[2];
#pragma unroll
        for (int mi = 0; mi < 2; mi++) {
            const __bf16* rp = &sA[(wm0 + mi * 16 + lr) * LDA + lhi * 8];
#pragma unroll
            for (int e = 0; e < 8; e++) a[mi][e] = rp[e];
#pragma unroll
            for (int e = 0; e < 8; e++) a[mi][8 + e] = rp[16 + e];
        }
#pragma unroll
        for (int nj = 0; nj < 2; nj++) {
            const __bf16* rp = &sB[(wn0 + nj * 16 + lr) * LDA + lhi * 8];
#pragma unroll
            for (int e = 0; e < 8; e++) bf[nj][e] = rp[e];
#pragma unroll
            for (int e = 0; e < 8; e++) bf[nj][8 + e] = rp[16 + e];
        }
#pragma unroll
        for (int mi = 0; mi < 2; mi++)
#pragma unroll
            for (int nj = 0; nj < 2; nj++)
                acc[mi][nj] = __builtin_amdgcn_wmma_f32_16x16x32_bf16(
                    false, a[mi], false, bf[nj], (short)0, acc[mi][nj], false, false);
    }

    // ---- epilogue: base pointer + constant r*P offsets (fold to ioffset)
#pragma unroll
    for (int mi = 0; mi < 2; mi++) {
        int mrow = mBase + wm0 + mi * 16 + lhi * 8;
#pragma unroll
        for (int nj = 0; nj < 2; nj++) {
            int ncol = nBase + wn0 + nj * 16 + lr;
            OutT* yp = Yb + (size_t)mrow * P + ncol;
#pragma unroll
            for (int r = 0; r < 8; r++)
                yp[(size_t)r * P] = (OutT)acc[mi][nj][r];
        }
    }
}

// ============================================================
// Depthwise 3x3, pad=1, bf16 in/out (streaming, traffic halved)
// ============================================================
__global__ __launch_bounds__(256)
void dwconv3x3(const __bf16* __restrict__ in, const float* __restrict__ w,
               __bf16* __restrict__ out, int total, int CH)
{
    int idx = blockIdx.x * 256 + threadIdx.x;
    if (idx >= total) return;
    int wq = idx % WDIM;
    int t  = idx / WDIM;
    int hq = t % HDIM;
    int plane = t / HDIM;
    int c = plane % CH;
    const float*  wp = w + c * 9;
    const __bf16* ip = in + (size_t)plane * HW;
    float s = 0.0f;
#pragma unroll
    for (int dy = 0; dy < 3; dy++) {
        int hy = hq + dy - 1;
        if (hy < 0 || hy >= HDIM) continue;
#pragma unroll
        for (int dx = 0; dx < 3; dx++) {
            int wx = wq + dx - 1;
            if (wx < 0 || wx >= WDIM) continue;
            s += (float)ip[(size_t)hy * WDIM + wx] * wp[dy * 3 + dx];
        }
    }
    out[idx] = (__bf16)s;
}

// ============================================================
// Windowed channel attention, one wave per (batch, window, head).
// qkv in bf16, out bf16. Two WMMA passes per head.
// ============================================================
#define AS 72   // LDS row stride (144 B, 16B aligned)

__global__ __launch_bounds__(32)
void win_channel_attn(const __bf16* __restrict__ qkv,  // [b][3C][H][W] bf16
                      const float* __restrict__ temp,  // [HEADS]
                      __bf16* __restrict__ out)        // [b][C][H][W] bf16
{
    __shared__ __bf16 sQ[48 * AS];  // q as A-matrix; reused for attn
    __shared__ __bf16 sK[48 * AS];  // k rows (B^T for pass 1)
    __shared__ __bf16 sV[64 * AS];  // v transposed [s][e] (B^T for pass 2)

    const int win  = blockIdx.x;
    const int head = blockIdx.y;
    const int bb   = blockIdx.z;
    const int wh = win >> 5, ww = win & 31;
    const int ln = threadIdx.x;
    const int lr = ln & 15, lhi = ln >> 4;

    const size_t pix0 = (size_t)(wh * PS) * WDIM + (size_t)(ww * PS);
    const __bf16* qp = qkv + ((size_t)bb * 3 * CDIM + head * DHEAD) * HW + pix0;
    const __bf16* kp = qp + (size_t)CDIM * HW;
    const __bf16* vp = qp + (size_t)2 * CDIM * HW;

    for (int row = ln; row < 64; row += 32)
#pragma unroll
        for (int e = 0; e < 64; e++) sV[row * AS + e] = (__bf16)0.0f;

    // load + L2-normalize q,k rows (lane owns rows ln, ln+32)
    for (int row = ln; row < 48; row += 32) {
        const __bf16* qr = qp + (size_t)row * HW;
        const __bf16* kr = kp + (size_t)row * HW;
        float ssq = 0.0f, ssk = 0.0f;
#pragma unroll
        for (int rr = 0; rr < PS; rr++)
#pragma unroll
            for (int cc = 0; cc < PS; cc++) {
                float a = (float)qr[rr * WDIM + cc]; ssq += a * a;
                float b = (float)kr[rr * WDIM + cc]; ssk += b * b;
            }
        float iq = 1.0f / fmaxf(sqrtf(ssq), 1e-12f);
        float ik = 1.0f / fmaxf(sqrtf(ssk), 1e-12f);
#pragma unroll
        for (int rr = 0; rr < PS; rr++)
#pragma unroll
            for (int cc = 0; cc < PS; cc++) {
                int s = rr * PS + cc;
                sQ[row * AS + s] = (__bf16)((float)qr[rr * WDIM + cc] * iq);
                sK[row * AS + s] = (__bf16)((float)kr[rr * WDIM + cc] * ik);
            }
#pragma unroll
        for (int s = WIN; s < 64; s++) {
            sQ[row * AS + s] = (__bf16)0.0f;
            sK[row * AS + s] = (__bf16)0.0f;
        }
    }
    __syncthreads();
    // v: straight bf16 copy, transposed [s][e]
    for (int row = ln; row < 48; row += 32) {
        const __bf16* vr = vp + (size_t)row * HW;
#pragma unroll
        for (int rr = 0; rr < PS; rr++)
#pragma unroll
            for (int cc = 0; cc < PS; cc++)
                sV[(rr * PS + cc) * AS + row] = vr[rr * WDIM + cc];
    }
    __syncthreads();

    // -------- pass 1: scores = q · k^T  (M=48, N=48, K=64) --------
    v8f acc1[3][3];
#pragma unroll
    for (int mi = 0; mi < 3; mi++)
#pragma unroll
        for (int nj = 0; nj < 3; nj++)
#pragma unroll
            for (int r = 0; r < 8; r++) acc1[mi][nj][r] = 0.0f;

#pragma unroll
    for (int ks = 0; ks < 64; ks += 32) {
        v16bf a[3], b[3];
#pragma unroll
        for (int mi = 0; mi < 3; mi++) {
            const __bf16* rp = &sQ[(mi * 16 + lr) * AS + ks + lhi * 8];
#pragma unroll
            for (int e = 0; e < 8; e++) a[mi][e] = rp[e];
#pragma unroll
            for (int e = 0; e < 8; e++) a[mi][8 + e] = rp[16 + e];
        }
#pragma unroll
        for (int nj = 0; nj < 3; nj++) {
            const __bf16* rp = &sK[(nj * 16 + lr) * AS + ks + lhi * 8];
#pragma unroll
            for (int e = 0; e < 8; e++) b[nj][e] = rp[e];
#pragma unroll
            for (int e = 0; e < 8; e++) b[nj][8 + e] = rp[16 + e];
        }
#pragma unroll
        for (int mi = 0; mi < 3; mi++)
#pragma unroll
            for (int nj = 0; nj < 3; nj++)
                acc1[mi][nj] = __builtin_amdgcn_wmma_f32_16x16x32_bf16(
                    false, a[mi], false, b[nj], (short)0, acc1[mi][nj], false, false);
    }

    // -------- temperature + row softmax (row = one 16-lane half) --------
    const float tau = temp[head];
#pragma unroll
    for (int mi = 0; mi < 3; mi++) {
#pragma unroll
        for (int r = 0; r < 8; r++) {
            float x0 = acc1[mi][0][r] * tau;
            float x1 = acc1[mi][1][r] * tau;
            float x2 = acc1[mi][2][r] * tau;
            float mx = fmaxf(x0, fmaxf(x1, x2));
#pragma unroll
            for (int off = 8; off; off >>= 1) mx = fmaxf(mx, __shfl_xor(mx, off, 16));
            float e0 = __expf(x0 - mx), e1 = __expf(x1 - mx), e2 = __expf(x2 - mx);
            float sm = e0 + e1 + e2;
#pragma unroll
            for (int off = 8; off; off >>= 1) sm += __shfl_xor(sm, off, 16);
            float inv = 1.0f / sm;
            int m = mi * 16 + lhi * 8 + r;
            sQ[m * AS + lr]      = (__bf16)(e0 * inv);
            sQ[m * AS + 16 + lr] = (__bf16)(e1 * inv);
            sQ[m * AS + 32 + lr] = (__bf16)(e2 * inv);
        }
    }
    __syncthreads();
    for (int row = ln; row < 48; row += 32)
#pragma unroll
        for (int s = 48; s < 64; s++) sQ[row * AS + s] = (__bf16)0.0f;
    __syncthreads();

    // -------- pass 2: out = attn · v  (M=48, N=64, K=64) --------
    v8f acc2[3][4];
#pragma unroll
    for (int mi = 0; mi < 3; mi++)
#pragma unroll
        for (int nj = 0; nj < 4; nj++)
#pragma unroll
            for (int r = 0; r < 8; r++) acc2[mi][nj][r] = 0.0f;

#pragma unroll
    for (int ks = 0; ks < 64; ks += 32) {
        v16bf a[3];
#pragma unroll
        for (int mi = 0; mi < 3; mi++) {
            const __bf16* rp = &sQ[(mi * 16 + lr) * AS + ks + lhi * 8];
#pragma unroll
            for (int e = 0; e < 8; e++) a[mi][e] = rp[e];
#pragma unroll
            for (int e = 0; e < 8; e++) a[mi][8 + e] = rp[16 + e];
        }
#pragma unroll
        for (int nj = 0; nj < 4; nj++) {
            v16bf b;
            const __bf16* rp = &sV[(nj * 16 + lr) * AS + ks + lhi * 8];
#pragma unroll
            for (int e = 0; e < 8; e++) b[e] = rp[e];
#pragma unroll
            for (int e = 0; e < 8; e++) b[8 + e] = rp[16 + e];
#pragma unroll
            for (int mi = 0; mi < 3; mi++)
                acc2[mi][nj] = __builtin_amdgcn_wmma_f32_16x16x32_bf16(
                    false, a[mi], false, b, (short)0, acc2[mi][nj], false, false);
        }
    }

    // -------- scatter (constant r*HW offsets fold into ioffset) --------
    __bf16* op = out + ((size_t)bb * CDIM + head * DHEAD) * HW + pix0;
#pragma unroll
    for (int nj = 0; nj < 4; nj++) {
        int n = nj * 16 + lr;
        if (n < WIN) {
            int rr = n / PS, cc = n % PS;
#pragma unroll
            for (int mi = 0; mi < 3; mi++) {
                int m0 = mi * 16 + lhi * 8;
                __bf16* opp = op + (size_t)m0 * HW + rr * WDIM + cc;
#pragma unroll
                for (int r = 0; r < 8; r++)
                    opp[(size_t)r * HW] = (__bf16)acc2[mi][nj][r];
            }
        }
    }
}

// ============================================================
extern "C" void kernel_launch(void* const* d_in, const int* in_sizes, int n_in,
                              void* d_out, int out_size, void* d_ws, size_t ws_size,
                              hipStream_t stream)
{
    const float* x      = (const float*)d_in[0];  // [2,384,224,224]
    const float* qkv_w  = (const float*)d_in[1];  // [1152,384]
    const float* dw_w   = (const float*)d_in[2];  // [1152,1,3,3]
    const float* temp   = (const float*)d_in[3];  // [8,1,1]
    const float* proj_w = (const float*)d_in[4];  // [384,384]
    float*       outp   = (float*)d_out;

    const int B  = 2;
    const int C3 = 3 * CDIM;                       // 1152

    // ---- workspace carve-up (all bf16, 256B-aligned regions) ----
    size_t xElems    = (size_t)B * CDIM * HW;      // 38,535,168
    size_t qkvwElems = (size_t)C3 * CDIM;          // 442,368
    size_t pwElems   = (size_t)CDIM * CDIM;        // 147,456
    size_t qkvElems  = (size_t)B * C3 * HW;        // 115,605,504

    __bf16* x_bf     = (__bf16*)d_ws;
    __bf16* qkvw_bf  = x_bf + xElems;
    __bf16* projw_bf = qkvw_bf + qkvwElems;
    __bf16* qkv_raw  = projw_bf + pwElems;
    __bf16* qkv_dw   = qkv_raw + qkvElems;
    __bf16* attn_o   = qkv_dw + qkvElems;

    // 0) one-time f32 -> bf16 conversions (input + weights)
    int n4x = (int)(xElems / 4);
    f32_to_bf16<<<(n4x + 255) / 256, 256, 0, stream>>>(x, x_bf, n4x);
    int n4w = (int)(qkvwElems / 4);
    f32_to_bf16<<<(n4w + 255) / 256, 256, 0, stream>>>(qkv_w, qkvw_bf, n4w);
    int n4p = (int)(pwElems / 4);
    f32_to_bf16<<<(n4p + 255) / 256, 256, 0, stream>>>(proj_w, projw_bf, n4p);

    // 1) qkv 1x1 projection -> bf16
    dim3 g1(HW / GT_N, C3 / GT_M, B);              // (784, 9, 2)
    gemm_wmma_bf16<__bf16><<<g1, 256, 0, stream>>>(qkvw_bf, x_bf, qkv_raw,
                                                   C3, CDIM, HW);
    // 2) depthwise 3x3 (bf16 -> bf16)
    int total = B * C3 * HW;
    dwconv3x3<<<(total + 255) / 256, 256, 0, stream>>>(qkv_raw, dw_w, qkv_dw,
                                                       total, C3);
    // 3) windowed channel attention (bf16 -> bf16)
    dim3 g3(NW * NW, HEADS, B);                    // (1024, 8, 2)
    win_channel_attn<<<g3, 32, 0, stream>>>(qkv_dw, temp, attn_o);

    // 4) output 1x1 projection -> f32
    dim3 g4(HW / GT_N, CDIM / GT_M, B);            // (784, 3, 2)
    gemm_wmma_bf16<float><<<g4, 256, 0, stream>>>(projw_bf, attn_o, outp,
                                                  CDIM, CDIM, HW);
}
